// SelfAttention_64347200029065
// MI455X (gfx1250) — matrix operand order repaired and checked by reading, not compiled, not run on hardware
//
#include <hip/hip_runtime.h>
#include <hip/hip_bf16.h>
#include <math.h>

// ---------------------------------------------------------------------------
// Self-attention block for MI455X (gfx1250), bf16 WMMA path.
// Ping-pong (unroll-by-2) software pipelining: no inter-buffer register moves,
// so no WMMA->VALU WAR hazard v_nops, and partial s_wait_loadcnt overlap.
//   x[B,T,D] @ w_in[D,3D] + b_in -> qkv ; 16-head attention (dh=64) ;
//   out = attn @ w_out + b_out
// B=2, T=2048, D=1024, H=16, dh=64
// ---------------------------------------------------------------------------

typedef __bf16 bf16_t;
typedef __attribute__((ext_vector_type(16))) __bf16 v16bf;
typedef __attribute__((ext_vector_type(8)))  __bf16 v8bf;
typedef __attribute__((ext_vector_type(8)))  float  v8f;

#define N_HEADS 16
#define D_EMBED 1024
#define D_HEAD  64
#define BB      2
#define TT      2048
#define MTOT    (BB * TT)      // 4096 tokens total

// ---- WMMA fragment loaders -------------------------------------------------
// A-matrix 16x32 bf16 layout (ISA 7.12.2): lane = (m = lane&15, h = lane>>4)
//   elements 0..7  <- K = k0 + 8h + 0..7
//   elements 8..15 <- K = k0 + 16 + 8h + 0..7
static __device__ __forceinline__ v16bf load_frag_a(const bf16_t* __restrict__ src,
                                                    int ld, int k0) {
  const int lane = threadIdx.x & 31;
  const int m = lane & 15, h = lane >> 4;
  const bf16_t* p = src + (size_t)m * ld + k0 + 8 * h;
  v8bf lo = *(const v8bf*)p;          // K = k0+8h .. +7
  v8bf hi = *(const v8bf*)(p + 16);   // K = k0+16+8h .. +7
  v16bf r;
#pragma unroll
  for (int i = 0; i < 8; ++i) { r[i] = lo[i]; r[i + 8] = hi[i]; }
  return r;
}

// B-matrix 32x16 bf16 layout: lane = (n = lane&15, h = lane>>4)
//   elements 0..7 <- K = k0 + 8h + e, elements 8..15 <- K = k0 + 16 + 8h + (e-8)   (srcT is [N][K] row-major, i.e. B^T)
static __device__ __forceinline__ v16bf load_frag_b(const bf16_t* __restrict__ srcT,
                                                    int ld, int k0) {
  const int lane = threadIdx.x & 31;
  const int n = lane & 15, h = lane >> 4;
  const bf16_t* p = srcT + (size_t)n * ld + k0 + 8 * h;
  v8bf lo = *(const v8bf*)p;
  v8bf hi = *(const v8bf*)(p + 16);
  v16bf r;
#pragma unroll
  for (int i = 0; i < 8; ++i) { r[i] = lo[i]; r[i + 8] = hi[i]; }
  return r;
}

static __device__ __forceinline__ v8f wmma_bf16(v16bf a, v16bf b, v8f c) {
  // 8 args: (neg_a, A, neg_b, B, c_mod, C, reuse_a, reuse_b)
  return __builtin_amdgcn_wmma_f32_16x16x32_bf16(false, a, false, b,
                                                 (short)0, c, false, false);
}

// ---- precision conversion / layout kernels ---------------------------------
__global__ void cvt_bf16_kernel(const float* __restrict__ in,
                                bf16_t* __restrict__ out, int n) {
  int i = blockIdx.x * blockDim.x + threadIdx.x;
  if (i < n) out[i] = (bf16_t)in[i];
}

// in: [R][C] row-major f32  ->  out: [C][R] row-major bf16 (transpose+convert)
__global__ void transpose_cvt_kernel(const float* __restrict__ in,
                                     bf16_t* __restrict__ out, int R, int C) {
  int i = blockIdx.x * blockDim.x + threadIdx.x;
  if (i < R * C) {
    int c = i / R, r = i % R;
    out[i] = (bf16_t)in[(size_t)r * C + c];
  }
}

// ---- shared GEMM core: one wave computes a 32x64 tile, K = D_EMBED ---------
struct GFrags { v16bf a0, a1, b[4]; };

static __device__ __forceinline__ void g_load(GFrags& f,
    const bf16_t* __restrict__ A0, const bf16_t* __restrict__ A1,
    const bf16_t* __restrict__ WT, int n0, int kk) {
  f.a0 = load_frag_a(A0, D_EMBED, kk);
  f.a1 = load_frag_a(A1, D_EMBED, kk);
#pragma unroll
  for (int j = 0; j < 4; ++j)
    f.b[j] = load_frag_b(WT + (size_t)(n0 + 16 * j) * D_EMBED, D_EMBED, kk);
}

static __device__ __forceinline__ void g_mma(v8f acc[2][4], const GFrags& f) {
#pragma unroll
  for (int j = 0; j < 4; ++j) acc[0][j] = wmma_bf16(f.a0, f.b[j], acc[0][j]);
#pragma unroll
  for (int j = 0; j < 4; ++j) acc[1][j] = wmma_bf16(f.a1, f.b[j], acc[1][j]);
}

static __device__ __forceinline__ void gemm_core(v8f acc[2][4],
    const bf16_t* __restrict__ A0, const bf16_t* __restrict__ A1,
    const bf16_t* __restrict__ WT, int n0, int m) {
  GFrags f0, f1;
  g_load(f0, A0, A1, WT, n0, 0);
  // ping-pong: load f1 while computing f0, load f0 while computing f1
  for (int kk = 0; kk < D_EMBED - 64; kk += 64) {
    __builtin_prefetch(A0 + (size_t)m * D_EMBED + kk + 128, 0, 3);
    g_load(f1, A0, A1, WT, n0, kk + 32);
    g_mma(acc, f0);
    g_load(f0, A0, A1, WT, n0, kk + 64);
    g_mma(acc, f1);
  }
  g_load(f1, A0, A1, WT, n0, D_EMBED - 32);
  g_mma(acc, f0);
  g_mma(acc, f1);
}

// ---- QKV projection: qkv = x @ w_in + b_in, scattered into head layouts ----
// grid: (MTOT/32, 3*D_EMBED/256), block: (32, 4).
__global__ __launch_bounds__(128) void qkv_gemm_kernel(
    const bf16_t* __restrict__ X,    // [MTOT][D_EMBED]
    const bf16_t* __restrict__ WT,   // [3*D_EMBED][D_EMBED]  (w_in^T)
    const float*  __restrict__ bias, // [3*D_EMBED]
    bf16_t* __restrict__ Qb,         // [B*H][T][64]
    bf16_t* __restrict__ Kb,         // [B*H][T][64]
    bf16_t* __restrict__ VTb) {      // [B*H][64][T]
  const int lane = threadIdx.x;
  const int wid  = threadIdx.y;
  const int row0 = blockIdx.x * 32;
  const int n0   = (blockIdx.y * 4 + wid) * 64;
  const int m = lane & 15, h = lane >> 4;

  const bf16_t* A0 = X + (size_t)row0 * D_EMBED;
  const bf16_t* A1 = X + (size_t)(row0 + 16) * D_EMBED;

  v8f acc[2][4] = {};
  gemm_core(acc, A0, A1, WT, n0, m);

  // Epilogue: value acc[i][j][v] at (row = row0+16i+8h+v, col = n0+16j+m)
  const int bidx = row0 >> 11;  // batch index (tile never crosses batches)
#pragma unroll
  for (int i = 0; i < 2; ++i) {
#pragma unroll
    for (int j = 0; j < 4; ++j) {
      const int n  = n0 + 16 * j + m;       // global output feature
      const float bv = bias[n];
      const int region = n >> 10;           // 0=Q, 1=K, 2=V
      const int nn = n & 1023;
      const int hh = nn >> 6;               // head
      const int dd = nn & 63;               // dim within head
      if (region == 2) {
        // V stored transposed: VT[(b*H+hh)][dd][t], t contiguous -> one b128 store
        v8bf pk;
#pragma unroll
        for (int v = 0; v < 8; ++v) pk[v] = (bf16_t)(acc[i][j][v] + bv);
        const int t0 = (row0 & 2047) + 16 * i + 8 * h;
        *(v8bf*)(VTb + ((size_t)(bidx * N_HEADS + hh) * D_HEAD + dd) * TT + t0) = pk;
      } else {
        bf16_t* dst = (region == 0) ? Qb : Kb;
#pragma unroll
        for (int v = 0; v < 8; ++v) {
          const int t = (row0 & 2047) + 16 * i + 8 * h + v;
          dst[((size_t)(bidx * N_HEADS + hh) * TT + t) * D_HEAD + dd] =
              (bf16_t)(acc[i][j][v] + bv);
        }
      }
    }
  }
}

// ---- Flash-attention: per wave one 16-query tile of one (b, head) ----------
// grid: (TT/64, N_HEADS, BB), block: (32, 4)
struct AFrags { v16bf k[4], v[4]; };

static __device__ __forceinline__ void a_load(AFrags& f,
    const bf16_t* __restrict__ Kh, const bf16_t* __restrict__ Vh, int k0) {
  // K fragments first, then V fragments: loads return in order, so the S WMMAs
  // drain only the K loads; V loads stay in flight across softmax + LDS.
  f.k[0] = load_frag_b(Kh + (size_t)k0 * D_HEAD, D_HEAD, 0);
  f.k[1] = load_frag_b(Kh + (size_t)k0 * D_HEAD, D_HEAD, 32);
  f.k[2] = load_frag_b(Kh + (size_t)(k0 + 16) * D_HEAD, D_HEAD, 0);
  f.k[3] = load_frag_b(Kh + (size_t)(k0 + 16) * D_HEAD, D_HEAD, 32);
#pragma unroll
  for (int j = 0; j < 4; ++j)
    f.v[j] = load_frag_b(Vh + (size_t)(16 * j) * TT, TT, k0);
}

__global__ __launch_bounds__(128) void attn_kernel(
    const bf16_t* __restrict__ Qb,   // [B*H][T][64]
    const bf16_t* __restrict__ Kb,   // [B*H][T][64]
    const bf16_t* __restrict__ VTb,  // [B*H][64][T]
    bf16_t* __restrict__ AO) {       // [MTOT][D_EMBED]
  __shared__ __align__(16) bf16_t ldsP[4][16][32];

  const int lane = threadIdx.x;
  const int wid  = threadIdx.y;
  const int m = lane & 15, h = lane >> 4;
  const int head = blockIdx.y;
  const int bb   = blockIdx.z;
  const int q0   = (blockIdx.x * 4 + wid) * 16;
  const int bh   = bb * N_HEADS + head;

  const bf16_t* Qh = Qb  + (size_t)bh * TT * D_HEAD;
  const bf16_t* Kh = Kb  + (size_t)bh * TT * D_HEAD;
  const bf16_t* Vh = VTb + (size_t)bh * D_HEAD * TT;

  const v16bf qa0 = load_frag_a(Qh + (size_t)q0 * D_HEAD, D_HEAD, 0);
  const v16bf qa1 = load_frag_a(Qh + (size_t)q0 * D_HEAD, D_HEAD, 32);

  float mrow[8], lrow[8];
  v8f o[4] = {};
#pragma unroll
  for (int v = 0; v < 8; ++v) { mrow[v] = -INFINITY; lrow[v] = 0.0f; }

  const float scale = 0.125f;  // 1/sqrt(64)

  // one key block (32 keys): S = Q@K^T, online softmax, O += P@V
  auto process = [&](const AFrags& f) {
    v8f s0 = {}, s1 = {};
    s0 = wmma_bf16(qa0, f.k[0], s0);
    s0 = wmma_bf16(qa1, f.k[1], s0);
    s1 = wmma_bf16(qa0, f.k[2], s1);
    s1 = wmma_bf16(qa1, f.k[3], s1);

    // Online softmax. Row (8h+v) is spread across the 16 lanes of this half,
    // so xor-shuffles with masks 1,2,4,8 reduce within the half.
#pragma unroll
    for (int v = 0; v < 8; ++v) {
      float a = s0[v] * scale, b = s1[v] * scale;
      float mx = fmaxf(a, b);
      mx = fmaxf(mx, __shfl_xor(mx, 1, 32));
      mx = fmaxf(mx, __shfl_xor(mx, 2, 32));
      mx = fmaxf(mx, __shfl_xor(mx, 4, 32));
      mx = fmaxf(mx, __shfl_xor(mx, 8, 32));
      const float mnew  = fmaxf(mrow[v], mx);
      const float alpha = __expf(mrow[v] - mnew);
      mrow[v] = mnew;
      const float pa = __expf(a - mnew);
      const float pb = __expf(b - mnew);
      float rs = pa + pb;
      rs += __shfl_xor(rs, 1, 32);
      rs += __shfl_xor(rs, 2, 32);
      rs += __shfl_xor(rs, 4, 32);
      rs += __shfl_xor(rs, 8, 32);
      lrow[v] = lrow[v] * alpha + rs;
#pragma unroll
      for (int j = 0; j < 4; ++j) o[j][v] *= alpha;
      // C-tile (row 8h+v, col m / 16+m) -> row-major P in LDS
      ldsP[wid][8 * h + v][m]      = (bf16_t)pa;
      ldsP[wid][8 * h + v][16 + m] = (bf16_t)pb;
    }
    __syncthreads();
    // Re-read P as an A-fragment (16x32 over the 32 keys)
    v16bf pfrag;
    {
      const bf16_t* pp = &ldsP[wid][m][8 * h];
      v8bf lo = *(const v8bf*)pp;
      v8bf hi = *(const v8bf*)(pp + 16);
#pragma unroll
      for (int i = 0; i < 8; ++i) { pfrag[i] = lo[i]; pfrag[i + 8] = hi[i]; }
    }
    __syncthreads();
    // O += P @ V : 4 d-tiles of 16 cols; V^T rows are contiguous over keys
#pragma unroll
    for (int j = 0; j < 4; ++j) o[j] = wmma_bf16(pfrag, f.v[j], o[j]);
  };

  // ping-pong over key blocks (TT/32 = 64 blocks, even)
  AFrags f0, f1;
  a_load(f0, Kh, Vh, 0);
  for (int kb = 0; kb < TT / 32; kb += 2) {
    a_load(f1, Kh, Vh, (kb + 1) * 32);
    process(f0);
    if (kb + 2 < TT / 32) a_load(f0, Kh, Vh, (kb + 2) * 32);
    process(f1);
  }

  // Normalize and write to attention-output buffer [MTOT][D_EMBED] (bf16)
#pragma unroll
  for (int j = 0; j < 4; ++j) {
#pragma unroll
    for (int v = 0; v < 8; ++v) {
      const int row = bb * TT + q0 + 8 * h + v;
      const int col = head * D_HEAD + 16 * j + m;
      AO[(size_t)row * D_EMBED + col] = (bf16_t)(o[j][v] / lrow[v]);
    }
  }
}

// ---- Output projection: out = attn @ w_out + b_out (fp32 result) -----------
// grid: (MTOT/32, D_EMBED/256), block: (32, 4).
__global__ __launch_bounds__(128) void out_gemm_kernel(
    const bf16_t* __restrict__ A,    // [MTOT][D_EMBED]
    const bf16_t* __restrict__ WT,   // [D_EMBED][D_EMBED] (w_out^T)
    const float*  __restrict__ bias, // [D_EMBED]
    float* __restrict__ C) {         // [MTOT][D_EMBED]
  const int lane = threadIdx.x;
  const int wid  = threadIdx.y;
  const int row0 = blockIdx.x * 32;
  const int n0   = (blockIdx.y * 4 + wid) * 64;
  const int m = lane & 15, h = lane >> 4;

  const bf16_t* A0 = A + (size_t)row0 * D_EMBED;
  const bf16_t* A1 = A + (size_t)(row0 + 16) * D_EMBED;

  v8f acc[2][4] = {};
  gemm_core(acc, A0, A1, WT, n0, m);

#pragma unroll
  for (int i = 0; i < 2; ++i) {
#pragma unroll
    for (int j = 0; j < 4; ++j) {
      const int n = n0 + 16 * j + m;
      const float bv = bias[n];
#pragma unroll
      for (int v = 0; v < 8; ++v) {
        const int row = row0 + 16 * i + 8 * h + v;
        C[(size_t)row * D_EMBED + n] = acc[i][j][v] + bv;
      }
    }
  }
}

// ---------------------------------------------------------------------------
extern "C" void kernel_launch(void* const* d_in, const int* in_sizes, int n_in,
                              void* d_out, int out_size, void* d_ws, size_t ws_size,
                              hipStream_t stream) {
  const float* x     = (const float*)d_in[0];  // [B,T,D]
  const float* w_in  = (const float*)d_in[1];  // [D,3D]
  const float* b_in  = (const float*)d_in[2];  // [3D]
  const float* w_out = (const float*)d_in[3];  // [D,D]
  const float* b_out = (const float*)d_in[4];  // [D]
  float* out = (float*)d_out;

  char* ws = (char*)d_ws;
  size_t off = 0;
  auto wsalloc = [&](size_t bytes) -> char* {
    char* p = ws + off;
    off += (bytes + 255) & ~(size_t)255;
    return p;
  };
  bf16_t* xb    = (bf16_t*)wsalloc((size_t)MTOT * D_EMBED * sizeof(bf16_t));
  bf16_t* winT  = (bf16_t*)wsalloc((size_t)3 * D_EMBED * D_EMBED * sizeof(bf16_t));
  bf16_t* woutT = (bf16_t*)wsalloc((size_t)D_EMBED * D_EMBED * sizeof(bf16_t));
  bf16_t* Qb    = (bf16_t*)wsalloc((size_t)MTOT * D_EMBED * sizeof(bf16_t));
  bf16_t* Kb    = (bf16_t*)wsalloc((size_t)MTOT * D_EMBED * sizeof(bf16_t));
  bf16_t* VTb   = (bf16_t*)wsalloc((size_t)MTOT * D_EMBED * sizeof(bf16_t));
  bf16_t* AOb   = (bf16_t*)wsalloc((size_t)MTOT * D_EMBED * sizeof(bf16_t));
  (void)ws_size; (void)in_sizes; (void)n_in; (void)out_size;

  {
    int n = MTOT * D_EMBED;
    cvt_bf16_kernel<<<(n + 255) / 256, 256, 0, stream>>>(x, xb, n);
  }
  {
    int n = D_EMBED * 3 * D_EMBED;
    transpose_cvt_kernel<<<(n + 255) / 256, 256, 0, stream>>>(w_in, winT,
                                                              D_EMBED, 3 * D_EMBED);
  }
  {
    int n = D_EMBED * D_EMBED;
    transpose_cvt_kernel<<<(n + 255) / 256, 256, 0, stream>>>(w_out, woutT,
                                                              D_EMBED, D_EMBED);
  }
  qkv_gemm_kernel<<<dim3(MTOT / 32, (3 * D_EMBED) / 256), dim3(32, 4), 0, stream>>>(
      xb, winT, b_in, Qb, Kb, VTb);
  attn_kernel<<<dim3(TT / 64, N_HEADS, BB), dim3(32, 4), 0, stream>>>(
      Qb, Kb, VTb, AOb);
  out_gemm_kernel<<<dim3(MTOT / 32, D_EMBED / 256), dim3(32, 4), 0, stream>>>(
      AOb, woutT, b_out, out);
}
